// ScaledDotProductAttention_10118942949548
// MI455X (gfx1250) — matrix operand order repaired and checked by reading, not compiled, not run on hardware
//
#include <hip/hip_runtime.h>

typedef __attribute__((ext_vector_type(16))) __bf16 v16bf;
typedef __attribute__((ext_vector_type(8)))  float  v8f;
typedef __attribute__((ext_vector_type(4)))  float  v4f;

#define S_LEN 2048
#define DK    64
#define NWAVE 4
#define NTHR  128

__global__ __launch_bounds__(NTHR) void sdpa_wmma_kernel(
    const float* __restrict__ Q,
    const float* __restrict__ K,
    const float* __restrict__ V,
    const unsigned char* __restrict__ mask,
    float* __restrict__ prob,
    float* __restrict__ attn)
{
    // 16 query rows x full key length, f32 scores/probs: 128 KB of LDS
    __shared__ float sc[16 * S_LEN];

    const int tid  = threadIdx.x;
    const int wave = tid >> 5;
    const int lane = tid & 31;
    const int half = lane >> 4;   // 0: lanes 0-15, 1: lanes 16-31
    const int l16  = lane & 15;

    const int bh = blockIdx.x / (S_LEN / 16);
    const int qt = blockIdx.x % (S_LEN / 16);
    const int q0 = qt * 16;

    const float* Qb = Q + ((size_t)bh * S_LEN + q0) * DK;
    const float* Kb = K + (size_t)bh * S_LEN * DK;
    const float* Vb = V + (size_t)bh * S_LEN * DK;
    const unsigned char* Mb = mask + ((size_t)bh * S_LEN + q0) * (size_t)S_LEN;
    float* Ob = prob + ((size_t)bh * S_LEN + q0) * DK;
    float* Ab = attn + ((size_t)bh * S_LEN + q0) * (size_t)S_LEN;

    // ---------------- Phase 1: scores = mask(QK^T / sqrt(dk)) -> LDS ----------------
    // A-fragment (Q tile, 16x64 split as two 16x32): row = l16,
    // element i -> d = 16*half + i
    v16bf qa0, qa1;
    {
        const float* qrow = Qb + (size_t)l16 * DK;
        const int s0 = 16 * half;
        const int s1 = 16 * half + 8;
#pragma unroll
        for (int i = 0; i < 8; ++i) {
            qa0[i]     = (__bf16)qrow[s0 + i];
            qa0[8 + i] = (__bf16)qrow[s1 + i];
            qa1[i]     = (__bf16)qrow[32 + s0 + i];
            qa1[8 + i] = (__bf16)qrow[32 + s1 + i];
        }
    }

    for (int j = wave; j < S_LEN / 16; j += NWAVE) {
        const int kk = j * 16;
        // B-fragment (K^T, 32x16): column n = l16, Kdim rows 16*half + i
        // b0[i] = K[kk + n][16*half + i], b1[i] = K[kk + n][32 + 16*half + i]
        v16bf b0, b1;
        const float* krow = Kb + (size_t)(kk + l16) * DK + 16 * half;
#pragma unroll
        for (int i = 0; i < 16; ++i) {
            b0[i] = (__bf16)krow[i];
            b1[i] = (__bf16)krow[32 + i];
        }
        v8f c = {};
        c = __builtin_amdgcn_wmma_f32_16x16x32_bf16(false, qa0, false, b0,
                                                    (short)0, c, false, false);
        c = __builtin_amdgcn_wmma_f32_16x16x32_bf16(false, qa1, false, b1,
                                                    (short)0, c, false, false);
        // C layout: VGPR r -> row M = r + 8*half, col N = l16
#pragma unroll
        for (int r = 0; r < 8; ++r) {
            const int M = r + 8 * half;
            const int N = l16;
            float s = c[r] * 0.125f;                       // 1/sqrt(64)
            // mask is stream-once (256 MB): non-temporal load, keep L2 for K/V
            const unsigned char mby =
                __builtin_nontemporal_load(Mb + (size_t)M * S_LEN + kk + N);
            if (mby) s = -1e9f;
            sc[M * S_LEN + kk + N] = s;
        }
    }
    __syncthreads();

    // ---------------- Phase 2: row softmax, write attn ----------------
    {
        const int r = tid >> 3;   // 0..15 query row
        const int g = tid & 7;    // 8 lanes cooperate per row
        float* row = sc + r * S_LEN;

        float mx = -3.402823e38f;
        for (int c4 = g; c4 < S_LEN / 4; c4 += 8) {
            v4f v = ((const v4f*)row)[c4];
            mx = fmaxf(mx, fmaxf(fmaxf(v.x, v.y), fmaxf(v.z, v.w)));
        }
        mx = fmaxf(mx, __shfl_xor(mx, 1));
        mx = fmaxf(mx, __shfl_xor(mx, 2));
        mx = fmaxf(mx, __shfl_xor(mx, 4));

        float sum = 0.f;
        for (int c4 = g; c4 < S_LEN / 4; c4 += 8) {
            v4f v = ((const v4f*)row)[c4];
            v.x = __expf(v.x - mx);
            v.y = __expf(v.y - mx);
            v.z = __expf(v.z - mx);
            v.w = __expf(v.w - mx);
            sum += v.x + v.y + v.z + v.w;
            ((v4f*)row)[c4] = v;
        }
        sum += __shfl_xor(sum, 1);
        sum += __shfl_xor(sum, 2);
        sum += __shfl_xor(sum, 4);
        const float inv = 1.f / sum;

        float* arow = Ab + (size_t)r * S_LEN;
        for (int c4 = g; c4 < S_LEN / 4; c4 += 8) {
            v4f v = ((v4f*)row)[c4];
            v.x *= inv; v.y *= inv; v.z *= inv; v.w *= inv;
            ((v4f*)row)[c4] = v;                       // normalized probs for phase 3
            // attn output is write-once streaming (1 GB): non-temporal b128 store
            __builtin_nontemporal_store(v, ((v4f*)arow) + c4);
        }
    }
    __syncthreads();

    // ---------------- Phase 3: O = P @ V (each wave owns 16 columns of D) ----------------
    {
        const int dcol0 = wave * 16;
        v8f o = {};
        for (int j = 0; j < S_LEN / 64; ++j) {
            const int kk = j * 64;
            // A-fragment from LDS probs: row = l16, k runs kk+16h+i
            v16bf a0, a1;
            const float* prow = sc + (size_t)l16 * S_LEN + kk;
            const int s0 = 16 * half;
            const int s1 = 16 * half + 8;
#pragma unroll
            for (int i = 0; i < 8; ++i) {
                a0[i]     = (__bf16)prow[s0 + i];
                a0[8 + i] = (__bf16)prow[s1 + i];
                a1[i]     = (__bf16)prow[32 + s0 + i];
                a1[8 + i] = (__bf16)prow[32 + s1 + i];
            }
            // B-fragment from V: column n = l16 (output dim), Kdim rows 16*half + i
            v16bf b0, b1;
            const float* vcol = Vb + (size_t)(kk + 16 * half) * DK + dcol0 + l16;
#pragma unroll
            for (int i = 0; i < 16; ++i) {
                b0[i] = (__bf16)vcol[(size_t)i * DK];
                b1[i] = (__bf16)vcol[(size_t)(32 + i) * DK];
            }
            o = __builtin_amdgcn_wmma_f32_16x16x32_bf16(false, a0, false, b0,
                                                        (short)0, o, false, false);
            o = __builtin_amdgcn_wmma_f32_16x16x32_bf16(false, a1, false, b1,
                                                        (short)0, o, false, false);
        }
#pragma unroll
        for (int r = 0; r < 8; ++r) {
            const int M = r + 8 * half;
            Ob[(size_t)M * DK + dcol0 + l16] = o[r];
        }
    }
}

extern "C" void kernel_launch(void* const* d_in, const int* in_sizes, int n_in,
                              void* d_out, int out_size, void* d_ws, size_t ws_size,
                              hipStream_t stream) {
    (void)in_sizes; (void)n_in; (void)d_ws; (void)ws_size; (void)out_size;
    const float* Q = (const float*)d_in[0];
    const float* K = (const float*)d_in[1];
    const float* V = (const float*)d_in[2];
    const unsigned char* mask = (const unsigned char*)d_in[3];

    float* prob = (float*)d_out;
    float* attn = prob + (size_t)4 * 16 * S_LEN * DK;   // prob first, then attn

    const int nblocks = 4 * 16 * (S_LEN / 16);          // B*H*(S/16) = 8192
    sdpa_wmma_kernel<<<nblocks, NTHR, 0, stream>>>(Q, K, V, mask, prob, attn);
}
